// DUConv2d_40664750358772
// MI455X (gfx1250) — matrix operand order repaired and checked by reading, not compiled, not run on hardware
//
#include <hip/hip_runtime.h>
#include <hip/hip_bf16.h>

// DuQuant conv2d for MI455X (gfx1250, wave32, WMMA).
// Pipeline: [init minmax+zeropage] -> [act rotate+minmax] -> [weight quant -> f16]
//           -> [act quant -> f16 NHWC] -> [implicit-GEMM conv, double-buffered
//              GLOBAL_LOAD_ASYNC_TO_LDS_B128 staging + v_wmma_f32_16x16x32_f16]

typedef __attribute__((ext_vector_type(16))) _Float16 v16h;
typedef __attribute__((ext_vector_type(8)))  float    v8f;
typedef __attribute__((ext_vector_type(4)))  unsigned uint4v;
typedef int v4i_ __attribute__((vector_size(16)));   // matches builtin's V4i param

#define BATCH 16
#define CIN   256
#define HH_   64
#define WW_   64
#define OUTC  256
#define KK    3
#define NROWS (BATCH * HH_ * WW_)   // 65536
#define KRED  (KK * KK * CIN)       // 2304
#define NSTEP (KRED / 32)           // 72
#define QMAX  15.0f

#if __has_builtin(__builtin_amdgcn_global_load_async_to_lds_b128)
#define HAVE_ASYNC_LDS 1
#else
#define HAVE_ASYNC_LDS 0
#endif

#define AS1CAST(p) ((__attribute__((address_space(1))) v4i_*)(p))
#define AS3CAST(p) ((__attribute__((address_space(3))) v4i_*)(p))

__device__ __forceinline__ void wait_async_le4() {
#if HAVE_ASYNC_LDS
#if __has_builtin(__builtin_amdgcn_s_wait_asynccnt)
  __builtin_amdgcn_s_wait_asynccnt(4);
#else
  asm volatile("s_wait_asynccnt 4" ::: "memory");
#endif
#endif
}
__device__ __forceinline__ void wait_async_le0() {
#if HAVE_ASYNC_LDS
#if __has_builtin(__builtin_amdgcn_s_wait_asynccnt)
  __builtin_amdgcn_s_wait_asynccnt(0);
#else
  asm volatile("s_wait_asynccnt 0" ::: "memory");
#endif
#endif
}

// ---- monotone float<->uint encoding so u32 atomicMin/Max give float min/max ----
__device__ __forceinline__ unsigned f2mono(float f) {
  unsigned u = __float_as_uint(f);
  return (u & 0x80000000u) ? ~u : (u | 0x80000000u);
}
__device__ __forceinline__ float mono2f(unsigned u) {
  u = (u & 0x80000000u) ? (u & 0x7fffffffu) : ~u;
  return __uint_as_float(u);
}

__global__ void k_init(unsigned* mm, unsigned* zb) {
  if (threadIdx.x == 0) { mm[0] = f2mono(0.0f); mm[1] = f2mono(0.0f); }
  if (threadIdx.x < 16) zb[threadIdx.x] = 0u;   // 64-byte zero page for padded rows
}

// ---------------------------------------------------------------------------
// Activation pass 1: rotate channels (blocks of 64), global min/max reduce.
// ---------------------------------------------------------------------------
__global__ __launch_bounds__(256)
void k_act_minmax(const float* __restrict__ x, const float* __restrict__ rot,
                  unsigned* __restrict__ mm) {
  __shared__ float sR[64 * 64];
  __shared__ float sX[4][CIN];
  __shared__ float sMin[256];
  __shared__ float sMax[256];
  const int tid = threadIdx.x;
  for (int i = tid; i < 64 * 64; i += 256) sR[i] = rot[i];
  const int c = tid, g = c >> 6, j = c & 63;
  const int row0 = blockIdx.x * 4;
  for (int r = 0; r < 4; ++r) {
    const int row = row0 + r;
    const int bb = row >> 12, hh = (row >> 6) & 63, wc = row & 63;
    sX[r][c] = x[(size_t)bb * (CIN * HH_ * WW_) + (size_t)c * (HH_ * WW_) + hh * WW_ + wc];
  }
  __syncthreads();
  float lmin = 0.0f, lmax = 0.0f;   // reference clamps range through 0
  for (int r = 0; r < 4; ++r) {
    const float* xs = &sX[r][g << 6];
    float acc = 0.0f;
#pragma unroll 8
    for (int b = 0; b < 64; ++b) acc = __builtin_fmaf(xs[b], sR[(b << 6) + j], acc);
    lmin = fminf(lmin, acc); lmax = fmaxf(lmax, acc);
  }
  sMin[tid] = lmin; sMax[tid] = lmax;
  __syncthreads();
  for (int s = 128; s > 0; s >>= 1) {
    if (tid < s) {
      sMin[tid] = fminf(sMin[tid], sMin[tid + s]);
      sMax[tid] = fmaxf(sMax[tid], sMax[tid + s]);
    }
    __syncthreads();
  }
  if (tid == 0) {
    atomicMin(&mm[0], f2mono(sMin[0]));
    atomicMax(&mm[1], f2mono(sMax[0]));
  }
}

// ---------------------------------------------------------------------------
// Weight quant: one block per output channel o. Rotate -> per-o minmax ->
// quant-dequant -> rotate back -> f16 B-matrix laid out [o][p*256 + c].
// ---------------------------------------------------------------------------
__global__ __launch_bounds__(256)
void k_weight_quant(const float* __restrict__ wgt, const float* __restrict__ rot,
                    _Float16* __restrict__ wq) {
  __shared__ float sR[64 * 64];
  __shared__ float sW[9][CIN];
  __shared__ float sQ[9][CIN];
  __shared__ float sMin[256];
  __shared__ float sMax[256];
  const int tid = threadIdx.x;
  const int o = blockIdx.x;
  for (int i = tid; i < 64 * 64; i += 256) sR[i] = rot[i];
  const int c = tid, g = c >> 6, j = c & 63;
  for (int p = 0; p < 9; ++p)
    sW[p][c] = wgt[(size_t)o * KRED + (size_t)c * 9 + p];   // OIHW -> [p][c]
  __syncthreads();
  float wr[9];
  float lmin = 0.0f, lmax = 0.0f;
  for (int p = 0; p < 9; ++p) {
    const float* xs = &sW[p][g << 6];
    float acc = 0.0f;
#pragma unroll 8
    for (int b = 0; b < 64; ++b) acc = __builtin_fmaf(xs[b], sR[(b << 6) + j], acc);
    wr[p] = acc;
    lmin = fminf(lmin, acc); lmax = fmaxf(lmax, acc);
  }
  sMin[tid] = lmin; sMax[tid] = lmax;
  __syncthreads();
  for (int s = 128; s > 0; s >>= 1) {
    if (tid < s) {
      sMin[tid] = fminf(sMin[tid], sMin[tid + s]);
      sMax[tid] = fmaxf(sMax[tid], sMax[tid + s]);
    }
    __syncthreads();
  }
  const float xmin = sMin[0], xmax = sMax[0];
  const float scale = fmaxf((xmax - xmin) / QMAX, 1e-8f);
  const float zp = rintf(-xmin / scale);
  for (int p = 0; p < 9; ++p) {
    float q = fminf(fmaxf(rintf(wr[p] / scale) + zp, 0.0f), QMAX);
    sQ[p][c] = (q - zp) * scale;
  }
  __syncthreads();
  for (int p = 0; p < 9; ++p) {
    const float* xs = &sQ[p][g << 6];
    float acc = 0.0f;
#pragma unroll 8
    for (int jj = 0; jj < 64; ++jj) acc = __builtin_fmaf(xs[jj], sR[(j << 6) + jj], acc);
    wq[(size_t)o * KRED + p * CIN + c] = (_Float16)acc;     // rotate back: R^T
  }
}

// ---------------------------------------------------------------------------
// Activation pass 2: recompute rotation, per-tensor quant-dequant, rotate back,
// store f16 NHWC for the implicit GEMM.
// ---------------------------------------------------------------------------
__global__ __launch_bounds__(256)
void k_act_quant(const float* __restrict__ x, const float* __restrict__ rot,
                 const unsigned* __restrict__ mm, _Float16* __restrict__ act) {
  __shared__ float sR[64 * 64];
  __shared__ float sX[4][CIN];
  __shared__ float sQ[4][CIN];
  const int tid = threadIdx.x;
  for (int i = tid; i < 64 * 64; i += 256) sR[i] = rot[i];
  const int c = tid, g = c >> 6, j = c & 63;
  const int row0 = blockIdx.x * 4;
  const float xmin = mono2f(mm[0]), xmax = mono2f(mm[1]);
  const float scale = fmaxf((xmax - xmin) / QMAX, 1e-8f);
  const float zp = rintf(-xmin / scale);
  for (int r = 0; r < 4; ++r) {
    const int row = row0 + r;
    const int bb = row >> 12, hh = (row >> 6) & 63, wc = row & 63;
    sX[r][c] = x[(size_t)bb * (CIN * HH_ * WW_) + (size_t)c * (HH_ * WW_) + hh * WW_ + wc];
  }
  __syncthreads();
  for (int r = 0; r < 4; ++r) {
    const float* xs = &sX[r][g << 6];
    float acc = 0.0f;
#pragma unroll 8
    for (int b = 0; b < 64; ++b) acc = __builtin_fmaf(xs[b], sR[(b << 6) + j], acc);
    float q = fminf(fmaxf(rintf(acc / scale) + zp, 0.0f), QMAX);
    sQ[r][c] = (q - zp) * scale;
  }
  __syncthreads();
  for (int r = 0; r < 4; ++r) {
    const float* xs = &sQ[r][g << 6];
    float acc = 0.0f;
#pragma unroll 8
    for (int jj = 0; jj < 64; ++jj) acc = __builtin_fmaf(xs[jj], sR[(j << 6) + jj], acc);
    act[(size_t)(row0 + r) * CIN + c] = (_Float16)acc;      // NHWC, row = b*4096+h*64+w
  }
}

// ---------------------------------------------------------------------------
// Implicit-GEMM conv, WMMA f16->f32, double-buffered async LDS staging.
//   M tile: one full W-line (64 rows, b/h fixed) -> trivial SAME padding.
//   N tile: 64 output channels (blockIdx.y).
//   K: 2304 = 9*256, stepped 32 (each step stays inside one (kh,kw)).
// 128 threads = 4 waves; wave w owns N cols [w*16,w*16+16) x all 64 M rows.
// Per stage each thread issues exactly 4 async b128 loads (padded rows pull
// from a zero page) so every wave's ASYNCcnt schedule is uniform.
// ---------------------------------------------------------------------------
struct Frag32B { union { uint4v u[2]; v16h h; }; };

__device__ __forceinline__ void stage_tile(
    const _Float16* __restrict__ act, const _Float16* __restrict__ wq,
    const _Float16* __restrict__ zpage,
    _Float16* __restrict__ Ad, _Float16* __restrict__ Bd,
    int b, int h, int n0, int lm, int lchunk, int step) {
  const int k0 = step << 5;
  const int p  = k0 >> 8;
  const int c0 = k0 & 255;
  const int kh = p / 3, kw = p - kh * 3;
  const int hsrc = h + kh - 1;
  const int wsrc = lm + kw - 1;
  const bool valid = ((unsigned)hsrc < (unsigned)HH_) & ((unsigned)wsrc < (unsigned)WW_);
  const _Float16* asrc = valid
      ? act + ((size_t)(((b << 6) + hsrc) << 6) + wsrc) * CIN + c0 + lchunk
      : zpage;                                     // 32 zero halves, 16B aligned
  const _Float16* bsrc = wq + (size_t)(n0 + lm) * KRED + k0 + lchunk;
  _Float16* adst = Ad + (lm << 5) + lchunk;
  _Float16* bdst = Bd + (lm << 5) + lchunk;
#if HAVE_ASYNC_LDS
  __builtin_amdgcn_global_load_async_to_lds_b128(AS1CAST(asrc),     AS3CAST(adst),     0, 0);
  __builtin_amdgcn_global_load_async_to_lds_b128(AS1CAST(asrc + 8), AS3CAST(adst + 8), 0, 0);
  __builtin_amdgcn_global_load_async_to_lds_b128(AS1CAST(bsrc),     AS3CAST(bdst),     0, 0);
  __builtin_amdgcn_global_load_async_to_lds_b128(AS1CAST(bsrc + 8), AS3CAST(bdst + 8), 0, 0);
#else
  uint4v a0 = *(const uint4v*)(asrc);
  uint4v a1 = *(const uint4v*)(asrc + 8);
  uint4v b0 = *(const uint4v*)(bsrc);
  uint4v b1 = *(const uint4v*)(bsrc + 8);
  __builtin_prefetch(bsrc + KRED, 0, 0);
  *(uint4v*)(adst)     = a0;
  *(uint4v*)(adst + 8) = a1;
  *(uint4v*)(bdst)     = b0;
  *(uint4v*)(bdst + 8) = b1;
#endif
}

__global__ __launch_bounds__(128)
void k_conv_wmma(const _Float16* __restrict__ act, const _Float16* __restrict__ wq,
                 const float* __restrict__ bias, const _Float16* __restrict__ zpage,
                 float* __restrict__ out) {
  __shared__ __align__(16) _Float16 As[2][64 * 32];   // [buf][m][k]
  __shared__ __align__(16) _Float16 Bs[2][64 * 32];   // [buf][n][k]
  const int tid  = threadIdx.x;
  const int wave = tid >> 5, lane = tid & 31;
  const int bh = blockIdx.x;            // b*64 + h
  const int b  = bh >> 6, h = bh & 63;
  const int n0 = blockIdx.y << 6;       // output-channel tile base

  v8f acc[4] = {};

  const int lm     = tid >> 1;          // 0..63 : A row (w index) / B row (n)
  const int lchunk = (tid & 1) << 4;    // 0 or 16 halves (32B)

  // fragment addressing (this code's contiguous k = 16h+s order): two contiguous 16B chunks per fragment
  const int laneN = lane & 15;
  const int kbB   = (lane >> 4) << 4;   // B: halves [kbB, kbB+16)
  const int khA   = (lane >> 4) << 4;   // A: halves [khA, khA+16)

  stage_tile(act, wq, zpage, As[0], Bs[0], b, h, n0, lm, lchunk, 0);
  stage_tile(act, wq, zpage, As[1], Bs[1], b, h, n0, lm, lchunk, 1);

  for (int step = 0; step < NSTEP; ++step) {
    const int cur = step & 1;
    if (step + 1 < NSTEP) wait_async_le4();   // current buffer landed, next in flight
    else                  wait_async_le0();
    __syncthreads();

    const _Float16* Bb = &Bs[cur][(((wave << 4) + laneN) << 5) + kbB];
    Frag32B bf;
    bf.u[0] = *(const uint4v*)(Bb);
    bf.u[1] = *(const uint4v*)(Bb + 8);
#pragma unroll
    for (int i = 0; i < 4; ++i) {
      const _Float16* Aa = &As[cur][(((i << 4) + laneN) << 5) + khA];
      Frag32B af;
      af.u[0] = *(const uint4v*)(Aa);
      af.u[1] = *(const uint4v*)(Aa + 8);
      acc[i] = __builtin_amdgcn_wmma_f32_16x16x32_f16(
          false, af.h, false, bf.h, (short)0, acc[i], false, false);
    }
    __syncthreads();                          // everyone done reading buf `cur`
    if (step + 2 < NSTEP)
      stage_tile(act, wq, zpage, As[cur], Bs[cur], b, h, n0, lm, lchunk, step + 2);
  }

  // epilogue: C layout lanes 0-15 N=lane/M=v, lanes 16-31 M=8+v; fuse bias
  const int o   = n0 + (wave << 4) + laneN;
  const int mof = (lane >> 4) << 3;
  const float bv = bias[o];
#pragma unroll
  for (int i = 0; i < 4; ++i) {
#pragma unroll
    for (int v = 0; v < 8; ++v) {
      const int w = (i << 4) + mof + v;
      out[((size_t)(((b << 8) + o) << 6) + h) * WW_ + w] = acc[i][v] + bv;
    }
  }
}

extern "C" void kernel_launch(void* const* d_in, const int* in_sizes, int n_in,
                              void* d_out, int out_size, void* d_ws, size_t ws_size,
                              hipStream_t stream) {
  const float* x    = (const float*)d_in[0];   // [16,256,64,64]
  const float* wgt  = (const float*)d_in[1];   // [256,256,3,3]
  const float* bias = (const float*)d_in[2];   // [256]
  const float* rot  = (const float*)d_in[3];   // [64,64]
  float* out = (float*)d_out;

  char* ws = (char*)d_ws;
  const size_t actBytes = (size_t)NROWS * CIN * 2;   // 32 MiB
  const size_t wqBytes  = (size_t)OUTC * KRED * 2;   // 1.125 MiB
  _Float16* act   = (_Float16*)ws;
  _Float16* wq    = (_Float16*)(ws + actBytes);
  unsigned* mm    = (unsigned*)(ws + actBytes + wqBytes);
  _Float16* zpage = (_Float16*)(ws + actBytes + wqBytes + 64);   // 64B zero page

  k_init<<<1, 64, 0, stream>>>(mm, (unsigned*)zpage);
  k_act_minmax<<<NROWS / 4, 256, 0, stream>>>(x, rot, mm);
  k_weight_quant<<<OUTC, 256, 0, stream>>>(wgt, rot, wq);
  k_act_quant<<<NROWS / 4, 256, 0, stream>>>(x, rot, mm, act);

  dim3 grid(BATCH * HH_, OUTC / 64);
  k_conv_wmma<<<grid, 128, 0, stream>>>(act, wq, bias, zpage, out);
}
